// MyConv2d_18176301596710
// MI455X (gfx1250) — matrix operand order repaired and checked by reading, not compiled, not run on hardware
//
#include <hip/hip_runtime.h>

// ---------------------------------------------------------------------------
// 2D valid cross-correlation 4096x4096 (f32) * 31x31 (f32) -> 4066x4066 (f32)
// CDNA5 (gfx1250) wave32 WMMA, bf16x3 split (~f32 accuracy), Toeplitz-kernel
// formulation.  Each wave computes FOUR 16x16 output tiles so the
// (ky,chunk)-invariant kernel fragments are amortized over 12 WMMAs.
//
//   D[m][n] = sum_k A[m][k]*B[k][n]
//   A[m][k] = Ker[ky][cb + k - m]            (banded Toeplitz, tile-invariant)
//   B[k][n] = X[y0 + n + ky][x0 + cb + k]    (input tile rows, LDS bf16)
//   D[m][n] = out[y0 + n][x0 + m]
// ---------------------------------------------------------------------------

#define HW    4096
#define KW    31
#define OW    4066            // 4096 - 31 + 1

#define ROWDW 152             // dwords per staged LDS row (304 bf16 columns)
#define TROWS 62              // staged input rows per block (32 out + 30 halo)

typedef __attribute__((ext_vector_type(16))) __bf16 v16bf;
typedef __attribute__((ext_vector_type(8)))  float  v8f;
// dword-aligned float4 for result stores (HW accepts dword-aligned B128)
typedef float f4a4 __attribute__((ext_vector_type(4), aligned(4)));

union FragCast { uint4 u[2]; v16bf v; };

__device__ __forceinline__ unsigned bf16_rne_bits(float x) {
    unsigned u = __float_as_uint(x);
    return (u + 0x7FFFu + ((u >> 16) & 1u)) >> 16;
}

__device__ __forceinline__ void bf16_split(float x, unsigned& h, unsigned& l) {
    h = bf16_rne_bits(x);
    float hf = __uint_as_float(h << 16);
    l = bf16_rne_bits(x - hf);
}

// ---------------------------------------------------------------------------
// Pre-kernel: expand the 31x31 kernel into Toeplitz A-matrix WMMA fragments,
// stored in a 16x32 A-fragment lane/VGPR layout with contiguous k order:
//   lanes 0-15 : M = lane,    VGPR j: K = 2j,2j+1 (j = 0..7)
//   lanes 16-31: M = lane-16, same but K += 16
// Workspace layout: [part(hi/lo)][ky(31)][chunk(2)][lane(32)][dword(8)]
// Total 31744 dwords (~124 KB) -> L2-resident for the whole run.
// ---------------------------------------------------------------------------
__global__ void build_kfrags(const float* __restrict__ Kc,
                             unsigned* __restrict__ frag) {
    int idx = blockIdx.x * 256 + threadIdx.x;
    const int TOT = 2 * 31 * 2 * 32 * 8;
    if (idx >= TOT) return;

    int dw   = idx & 7;        int t = idx >> 3;
    int lane = t & 31;         t >>= 5;
    int cb   = (t & 1) * 32;   t >>= 1;
    int ky   = t % 31;
    int part = t / 31;         // 0 = hi, 1 = lo

    int m = lane & 15;
    int q = lane >> 4;
    int kbase = dw * 2 + 16 * q;

    unsigned out = 0;
    #pragma unroll
    for (int e = 0; e < 2; ++e) {
        int c  = cb + kbase + e;
        int kx = c - m;
        float v = (kx >= 0 && kx < KW) ? Kc[ky * KW + kx] : 0.0f;
        unsigned h, l;
        bf16_split(v, h, l);
        out |= (part ? l : h) << (16 * e);
    }
    frag[idx] = out;
}

// ---------------------------------------------------------------------------
// Main kernel: block = 256 threads = 8 wave32s covering 32 rows x 256 cols.
// Wave w: ry = w>>2, cx = w&3; four tiles at xloc = cx*16 + 64*t, t=0..3.
// ---------------------------------------------------------------------------
__global__ __launch_bounds__(256)
void conv_wmma(const float* __restrict__ X,
               const unsigned* __restrict__ kfrag,
               float* __restrict__ Y) {
    __shared__ __align__(16) unsigned ldsHi[TROWS * ROWDW];
    __shared__ __align__(16) unsigned ldsLo[TROWS * ROWDW];

    const int tid = threadIdx.x;
    const int gx0 = blockIdx.x * 256;
    const int gy0 = blockIdx.y * 32;

    // Hint the (tiny, hot) fragment table into cache early.
    __builtin_prefetch(kfrag, 0, 3);

    // ---- stage input tile: f32 -> packed bf16 hi/lo pairs in LDS ----------
    if (gx0 + 2 * ROWDW <= HW && gy0 + TROWS <= HW) {
        // Interior fast path: vectorized float4 loads, ds_store_b64 stores.
        for (int i = tid; i < TROWS * (ROWDW / 2); i += 256) {
            int r = i / (ROWDW / 2), c4 = i - r * (ROWDW / 2);
            const float4 x =
                *(const float4*)&X[(size_t)(gy0 + r) * HW + gx0 + 4 * c4];
            unsigned h0, l0, h1, l1, h2, l2, h3, l3;
            bf16_split(x.x, h0, l0);
            bf16_split(x.y, h1, l1);
            bf16_split(x.z, h2, l2);
            bf16_split(x.w, h3, l3);
            uint2 hp, lp;
            hp.x = h0 | (h1 << 16); hp.y = h2 | (h3 << 16);
            lp.x = l0 | (l1 << 16); lp.y = l2 | (l3 << 16);
            *(uint2*)&ldsHi[r * ROWDW + 2 * c4] = hp;
            *(uint2*)&ldsLo[r * ROWDW + 2 * c4] = lp;
        }
    } else {
        // Boundary path: scalar clamped loads (clamped texels only ever
        // multiply the zero pad of the Toeplitz band / dead outputs).
        for (int i = tid; i < TROWS * ROWDW; i += 256) {
            int r = i / ROWDW, d = i - r * ROWDW;
            int gr = gy0 + r;      gr = gr > (HW - 1) ? (HW - 1) : gr;
            int c0 = gx0 + 2 * d;  c0 = c0 > (HW - 1) ? (HW - 1) : c0;
            int c1 = gx0 + 2 * d + 1;
            c1 = c1 > (HW - 1) ? (HW - 1) : c1;
            float x0 = X[(size_t)gr * HW + c0];
            float x1 = X[(size_t)gr * HW + c1];
            unsigned h0, l0, h1, l1;
            bf16_split(x0, h0, l0);
            bf16_split(x1, h1, l1);
            ldsHi[i] = h0 | (h1 << 16);
            ldsLo[i] = l0 | (l1 << 16);
        }
    }
    __syncthreads();

    const int wave = tid >> 5;          // wave32
    const int lane = tid & 31;
    const int ry = wave >> 2, cx = wave & 3;
    const int n = lane & 15;            // B column / D column index N
    const int q = lane >> 4;            // lane half

    v8f acc0 = {}, acc1 = {}, acc2 = {}, acc3 = {};

    const uint4* kf = (const uint4*)kfrag;
    const int kfLane2 = lane * 2;
    const int bcol = cx * 8 + 8 * q;    // dword column base within an LDS row

    for (int ky = 0; ky < 31; ++ky) {
        const int rowBase = (ry * 16 + n + ky) * ROWDW;
        #pragma unroll
        for (int cs = 0; cs < 2; ++cs) {
            // A (kernel Toeplitz) fragments: tile-invariant, loaded once,
            // fed to 12 WMMAs below.
            int baseH = (((0 * 31 + ky) * 2 + cs) * 32) * 2 + kfLane2;
            int baseL = (((1 * 31 + ky) * 2 + cs) * 32) * 2 + kfLane2;
            FragCast Ah, Al;
            Ah.u[0] = kf[baseH];     Ah.u[1] = kf[baseH + 1];
            Al.u[0] = kf[baseL];     Al.u[1] = kf[baseL + 1];

            const int dbase = rowBase + bcol + cs * 16;

            #pragma unroll
            for (int t = 0; t < 4; ++t) {
                // B (input) fragment: 8 consecutive 16B-aligned dwords
                const int dcol = dbase + 32 * t;
                const uint4* bh = (const uint4*)&ldsHi[dcol];
                const uint4* bl = (const uint4*)&ldsLo[dcol];
                FragCast Bh, Bl;
                Bh.u[0] = bh[0]; Bh.u[1] = bh[1];
                Bl.u[0] = bl[0]; Bl.u[1] = bl[1];

                v8f a = (t == 0) ? acc0 : (t == 1) ? acc1 : (t == 2) ? acc2 : acc3;
                // bf16x3: hi*hi + hi*lo + lo*hi, f32 accumulate
                a = __builtin_amdgcn_wmma_f32_16x16x32_bf16(
                    false, Ah.v, false, Bh.v, (short)0, a, false, false);
                a = __builtin_amdgcn_wmma_f32_16x16x32_bf16(
                    false, Al.v, false, Bh.v, (short)0, a, false, false);
                a = __builtin_amdgcn_wmma_f32_16x16x32_bf16(
                    false, Ah.v, false, Bl.v, (short)0, a, false, false);
                if      (t == 0) acc0 = a;
                else if (t == 1) acc1 = a;
                else if (t == 2) acc2 = a;
                else             acc3 = a;
            }
        }
    }

    // ---- store: D[m][n] = out[y0+n][x0+m]; lane's 8 results per tile are
    //      8 consecutive columns (m = r + 8q) in row y0+n ------------------
    const int row = gy0 + ry * 16 + n;
    if (row < OW) {
        #pragma unroll
        for (int t = 0; t < 4; ++t) {
            const v8f a = (t == 0) ? acc0 : (t == 1) ? acc1 : (t == 2) ? acc2 : acc3;
            const int col0 = gx0 + cx * 16 + 64 * t + 8 * q;
            if (col0 + 7 < OW) {
                float* yp = &Y[(size_t)row * OW + col0];
                f4a4 v0 = { a[0], a[1], a[2], a[3] };
                f4a4 v1 = { a[4], a[5], a[6], a[7] };
                *(f4a4*)yp       = v0;    // dword-aligned b128 stores
                *(f4a4*)(yp + 4) = v1;
            } else {
                #pragma unroll
                for (int r = 0; r < 8; ++r) {
                    int col = col0 + r;
                    if (col < OW) Y[(size_t)row * OW + col] = a[r];
                }
            }
        }
    }
}

// ---------------------------------------------------------------------------
extern "C" void kernel_launch(void* const* d_in, const int* in_sizes, int n_in,
                              void* d_out, int out_size, void* d_ws, size_t ws_size,
                              hipStream_t stream) {
    const float* X  = (const float*)d_in[0];   // 4096*4096 f32
    const float* Kc = (const float*)d_in[1];   // 31*31 f32
    float* Y = (float*)d_out;                  // 4066*4066 f32
    unsigned* frag = (unsigned*)d_ws;          // 31744 dwords (~124 KB)

    build_kfrags<<<248, 256, 0, stream>>>(Kc, frag);

    dim3 grid((OW + 255) / 256, (OW + 31) / 32); // 16 x 128 blocks
    conv_wmma<<<grid, 256, 0, stream>>>(X, frag, Y);
}
